// LegoCNN_4363686773276
// MI455X (gfx1250) — matrix operand order repaired and checked by reading, not compiled, not run on hardware
//
#include <hip/hip_runtime.h>
#include <hip/hip_bf16.h>
#include <stdint.h>

// ---------------------------------------------------------------------------
// B=128, INA=128, H=32, OUT=512, K=3, LEGO=128, NUM_LABELS=10.
// Conv-as-GEMM per batch: M=LEGO(128), Kdim=INA*9=1152, N=H*H=1024.
// Second stage is 1-sparse: out[b,o,hw] = coef[o] * feat[b, sel[o], hw].
// ---------------------------------------------------------------------------
#define BATCH   128
#define INA     128
#define HH      32
#define NPIX    1024
#define OUTC    512
#define LEGO    128
#define KDIM    1152

typedef __bf16 bf16_t;
typedef bf16_t v16bf __attribute__((ext_vector_type(16)));
typedef float  v8f   __attribute__((ext_vector_type(8)));
typedef float  v4f   __attribute__((ext_vector_type(4)));   // clang vector: OK for nontemporal builtins

static __device__ __forceinline__ unsigned short f2bf(float f) {
    unsigned u = __float_as_uint(f);
    u += 0x7FFFu + ((u >> 16) & 1u);
    return (unsigned short)(u >> 16);
}

// ---------------------------------------------------------------------------
// Prep kernel (1 block, 512 threads): per output channel o,
//   sel[o]  = argmax_l comb0[o,l]   (first-max, matching jnp.argmax)
//   coef[o] = coeffs0[o, sel[o]]
// plus a CSR inverted index lego -> {o}:  offsets[129], items[512].
// ---------------------------------------------------------------------------
__global__ __launch_bounds__(OUTC)
void lego_prep_kernel(const float* __restrict__ coeffs0,
                      const float* __restrict__ comb0,
                      int* __restrict__ sel, float* __restrict__ coef,
                      int* __restrict__ offsets, int* __restrict__ items) {
    __shared__ int cnt[LEGO];
    __shared__ int base[LEGO + 1];

    const int o = threadIdx.x;                 // 0..511
    if (o < LEGO) cnt[o] = 0;
    __syncthreads();

    const float* row = comb0 + (size_t)o * LEGO;
    int   arg = 0;
    float mx  = row[0];
    for (int j = 1; j < LEGO; ++j) {
        float v = row[j];
        if (v > mx) { mx = v; arg = j; }
    }
    sel[o]  = arg;
    coef[o] = coeffs0[(size_t)o * LEGO + arg];
    atomicAdd(&cnt[arg], 1);
    __syncthreads();

    if (o == 0) {
        int s = 0;
        for (int l = 0; l < LEGO; ++l) { base[l] = s; s += cnt[l]; }
        base[LEGO] = s;
    }
    __syncthreads();

    if (o <= LEGO) offsets[o] = base[o];
    if (o < LEGO)  cnt[o] = 0;
    __syncthreads();

    int pos = base[arg] + atomicAdd(&cnt[arg], 1);
    items[pos] = o;
}

// ---------------------------------------------------------------------------
// Main fused kernel.  Grid: (nTile=8, mTile=8, b=128).  Block: 256 = 8 wave32.
// Tile: 16 legos x 128 pixels (4 image rows).  K' = r*128 + c (same
// permutation on A and B), so each K=32 WMMA step is one 3x3 tap over 32
// contiguous input channels.
// ---------------------------------------------------------------------------
__global__ __launch_bounds__(256)
void lego_conv_wmma_kernel(const float* __restrict__ x,
                           const int*   __restrict__ label,
                           const float* __restrict__ ff,       // [10][128][128][3][3]
                           const int*   __restrict__ sel,
                           const float* __restrict__ coef,
                           const int*   __restrict__ offsets,  // [129]
                           const int*   __restrict__ items,    // [512]
                           float*       __restrict__ out) {
    __shared__ unsigned short Wl[16 * KDIM];     // 36864 B : bf16 A tile [m][k']
    __shared__ unsigned short Xs[6 * 34 * 32];   // 13056 B : bf16 x slab [row][colpad][c]
    __shared__ float          featF[16 * 128];   //  8192 B : f32 feat tile [m][pix]

    const int nT   = blockIdx.x;
    const int mT   = blockIdx.y;
    const int b    = blockIdx.z;
    const int tid  = threadIdx.x;
    const int lane = tid & 31;
    const int wv   = tid >> 5;
    const int hi   = (lane >> 4) & 1;
    const int y0   = nT * 4;
    const int mBase = mT * 16;

    // ---- stage W tile: ff[lab][mBase+m][c][ky][kx] -> Wl[m][r*128+c] ----
    {
        const int lab = label[b];
        const float* Wg = ff + ((size_t)lab * LEGO + (size_t)mBase) * KDIM;
        for (int i = tid; i < 16 * KDIM; i += 256) {
            int m   = i / KDIM;
            int rem = i - m * KDIM;      // rem = c*9 + r (coalesced source order)
            int c   = rem / 9;
            int r   = rem - c * 9;
            Wl[m * KDIM + r * 128 + c] = f2bf(Wg[i]);
        }
    }

    // ---- fragment geometry (ISA 7.12.2, wave32 bf16) ----
    // A 16x32: lane m(0-15): K 0-15 ; lane m+16: K 16-31 (contiguous order chosen here, not the ISA table).
    const int aRow = (lane & 15) * KDIM;
    const int aOff = hi * 16;
    // B 32x16: lane n(0-15): K 0-15 ; lane n+16: K 16-31 (same contiguous order as A).  Column = pixel.
    const int n    = lane & 15;
    const int poff = wv * 16 + n;        // pixel within 128-pixel tile
    const int yl   = poff >> 5;
    const int xc   = poff & 31;
    const int bOff = hi * 16;

    v8f acc = {};
    const float* xb = x + (size_t)b * INA * NPIX;

    for (int cb = 0; cb < 4; ++cb) {
        __syncthreads();                 // protect Xs (and Wl on cb==0)

        // stage x slab: 32ch x 6 rows x 32 cols -> Xs[row][col+1][c] (bf16)
        for (int i = tid; i < 32 * 6 * 32; i += 256) {
            int xx = i & 31;
            int yy = (i >> 5) % 6;
            int c  = i / 192;
            int gy = y0 - 1 + yy;
            float v = (gy >= 0 && gy < HH)
                    ? xb[((size_t)(cb * 32 + c)) * NPIX + gy * HH + xx]
                    : 0.0f;
            Xs[(yy * 34 + xx + 1) * 32 + c] = f2bf(v);
        }
        for (int i = tid; i < 6 * 2 * 32; i += 256) {   // zero pad cols 0 & 33
            int c    = i & 31;
            int yy   = (i >> 5) % 6;
            int side = i / 192;
            Xs[(yy * 34 + side * 33) * 32 + c] = 0;
        }
        __syncthreads();

        #pragma unroll
        for (int r = 0; r < 9; ++r) {
            const int ky = r / 3, kx = r % 3;

            const uint4* ap = (const uint4*)&Wl[aRow + r * 128 + cb * 32 + aOff];
            uint4 a0 = ap[0];
            uint4 a1 = ap[1];

            const int row = yl + ky;
            const int col = xc + kx;
            const uint4* bp = (const uint4*)&Xs[(row * 34 + col) * 32 + bOff];
            uint4 b0 = bp[0];
            uint4 b1 = bp[1];

            union { uint4 u[2]; v16bf v; } ua, ub;
            ua.u[0] = a0; ua.u[1] = a1;
            ub.u[0] = b0; ub.u[1] = b1;

            acc = __builtin_amdgcn_wmma_f32_16x16x32_bf16(
                    false, ua.v, false, ub.v, (short)0, acc, false, false);
        }
    }

    // ---- dump accumulator tile to LDS: featF[m][pix] ----
    // C/D layout: lane L holds column N=L%16; VGPR v holds row M = v + 8*hi.
    #pragma unroll
    for (int v = 0; v < 8; ++v)
        featF[(v + 8 * hi) * 128 + poff] = acc[v];
    __syncthreads();

    // ---- sparse channel expansion: one wave per mapped output channel ----
    // CSR slice for this lego tile is contiguous: items[offs0 .. offs1).
    const int offs0 = offsets[mBase];
    const int offs1 = offsets[mBase + 16];
    const size_t outB = (size_t)b * OUTC * NPIX + (size_t)nT * 128 + (lane << 2);

    for (int idx = offs0 + wv; idx < offs1; idx += 8) {
        int   o  = items[idx];
        int   m  = sel[o] - mBase;       // in [0,16)
        float cf = coef[o];
        const v4f fv = *(const v4f*)&featF[m * 128 + (lane << 2)];
        v4f vv = fv * cf;
        // streamed output: non-temporal b128 store, 512B contiguous per wave
        __builtin_nontemporal_store(vv, (v4f*)&out[outB + (size_t)o * NPIX]);
    }
}

// ---------------------------------------------------------------------------
extern "C" void kernel_launch(void* const* d_in, const int* in_sizes, int n_in,
                              void* d_out, int out_size, void* d_ws, size_t ws_size,
                              hipStream_t stream) {
    const float* x      = (const float*)d_in[0];   // (128,128,32,32)
    const int*   label  = (const int*)  d_in[1];   // (128,)
    const float* ff     = (const float*)d_in[2];   // (10,128,128,3,3)
    const float* coeffs = (const float*)d_in[3];   // (4,512,128,1,1) -> slice 0
    const float* comb   = (const float*)d_in[4];   // (4,512,128,1,1) -> slice 0
    float*       out    = (float*)d_out;           // (128,512,32,32)

    char* ws = (char*)d_ws;
    int*   sel     = (int*)  (ws);                 //  512 * 4
    float* coef    = (float*)(ws + 2048);          //  512 * 4
    int*   offsets = (int*)  (ws + 4096);          //  129 * 4 (padded to 1KB)
    int*   items   = (int*)  (ws + 5120);          //  512 * 4

    lego_prep_kernel<<<dim3(1), dim3(OUTC), 0, stream>>>(coeffs, comb,
                                                         sel, coef, offsets, items);

    dim3 grid(8, 8, BATCH);   // (nTile, mTile, batch)
    lego_conv_wmma_kernel<<<grid, dim3(256), 0, stream>>>(
        x, label, ff, sel, coef, offsets, items, out);

    (void)in_sizes; (void)n_in; (void)out_size; (void)ws_size;
}
